// SBSM_1812476199143
// MI455X (gfx1250) — matrix operand order repaired and checked by reading, not compiled, not run on hardware
//
#include <hip/hip_runtime.h>
#include <hip/hip_bf16.h>
#include <math.h>

#define Bb   2
#define Cc   48
#define Hh   64
#define Ww   64
#define LL   (Hh*Ww)      // 4096
#define DIv  96
#define DSv  16
#define DCONVv 4
#define DTRv 3
#define HFv  96
#define XPN  (DTRv + 2*DSv)   // 35
#define CDIV(a,b) (((a)+(b)-1)/(b))

typedef __attribute__((ext_vector_type(16))) _Float16 v16h;
typedef __attribute__((ext_vector_type(8)))  float    v8f;

// ---------------------------------------------------------------------------
// WMMA GEMM:  C[M,N] = A[M,K] (row-major f32) * W[N,K]^T (row-major f32)
// N, K are compile-time so all range guards fold and loads vectorize to b128.
// One wave per 16x16 output tile, v_wmma_f32_16x16x32_f16, fp32 accumulate.
// Per-lane packing follows the CDNA5 ISA layouts:
//   A 16x32 f16 : lane<16 row=lane, halfs[0..7]=K(k0+0..7),  halfs[8..15]=K(k0+16..23)
//                 lane>=16 row=lane-16, halfs=K(k0+8..15), K(k0+24..31)
//   B 32x16 f16 : same K packing as A; lane<16 col=lane, lane>=16 col=lane-16
//   C/D f32     : lane<16 (M=v, N=lane); lane>=16 (M=8+v, N=lane-16)
// All K used here are multiples of 16, so within a 32-wide chunk the low half
// is always valid and the high half is valid unless it's the final 16-wide
// partial chunk (a compile-time condition).
// ---------------------------------------------------------------------------
template <int N, int K>
__global__ void wmma_gemm_nt(const float* __restrict__ A,
                             const float* __restrict__ Wt,
                             float* __restrict__ C, int M) {
  static_assert(K % 16 == 0, "K must be multiple of 16");
  constexpr int nTilesN = (N + 15) / 16;
  int wave = (int)((blockIdx.x * blockDim.x + threadIdx.x) >> 5);
  int lane = (int)(threadIdx.x & 31);
  int nTilesM = M >> 4;
  if (wave >= nTilesM * nTilesN) return;
  int tm = wave / nTilesN, tn = wave % nTilesN;
  int m0 = tm << 4, n0 = tn << 4;
  int half = lane >> 4, lid = lane & 15;
  int arow = m0 + lid;
  int bcol = n0 + lid;
  const bool bvalid = (N % 16 == 0) ? true : (bcol < N);
  const float* __restrict__ Ap = A + (size_t)arow * K + half * 8;
  const float* __restrict__ Bp = Wt + (size_t)(bvalid ? bcol : 0) * K + half * 8;

  v8f acc = {};
#pragma unroll
  for (int k0 = 0; k0 < K; k0 += 32) {
    v16h av = {};
    v16h bv = {};
    // A low half of chunk: always in range (K - k0 >= 16, K % 16 == 0)
#pragma unroll
    for (int i = 0; i < 8; ++i) av[i] = (_Float16)Ap[k0 + i];
    if (k0 + 16 < K) {  // compile-time per unrolled iteration
#pragma unroll
      for (int i = 0; i < 8; ++i) av[8 + i] = (_Float16)Ap[k0 + 16 + i];
    }
    // B: same K packing as A; low half always valid, high half only when the
    // chunk is full.
    if (bvalid) {
#pragma unroll
      for (int i = 0; i < 8; ++i) bv[i] = (_Float16)Bp[k0 + i];
      if (k0 + 16 < K) {
#pragma unroll
        for (int i = 0; i < 8; ++i) bv[8 + i] = (_Float16)Bp[k0 + 16 + i];
      }
    }
    acc = __builtin_amdgcn_wmma_f32_16x16x32_f16(false, av, false, bv,
                                                 (short)0, acc, false, false);
  }
  if (bvalid) {
#pragma unroll
    for (int v = 0; v < 8; ++v) {
      int mrow = m0 + half * 8 + v;
      C[(size_t)mrow * N + bcol] = acc[v];
    }
  }
}

// ---------------------------------------------------------------------------
// LayerNorm over channels of a (B,C,H*W) spatial tensor. out_layout:
//   0 -> spatial (B,C,HW), 1 -> token-major (B,HW,C)
// ---------------------------------------------------------------------------
__global__ void ln4d_kernel(const float* __restrict__ x,
                            const float* __restrict__ w,
                            const float* __restrict__ b,
                            float* __restrict__ out, int token_major) {
  int i = blockIdx.x * blockDim.x + threadIdx.x;
  if (i >= Bb * LL) return;
  int bb = i / LL, p = i % LL;
  float v[Cc]; float mu = 0.f;
#pragma unroll
  for (int c = 0; c < Cc; ++c) { v[c] = x[((size_t)bb*Cc + c)*LL + p]; mu += v[c]; }
  mu *= (1.f / Cc);
  float var = 0.f;
#pragma unroll
  for (int c = 0; c < Cc; ++c) { float d = v[c] - mu; var += d*d; }
  var *= (1.f / Cc);
  float rs = rsqrtf(var + 1e-5f);
#pragma unroll
  for (int c = 0; c < Cc; ++c) {
    float o = (v[c] - mu) * rs * w[c] + b[c];
    if (token_major) out[((size_t)bb*LL + p)*Cc + c] = o;
    else             out[((size_t)bb*Cc + c)*LL + p] = o;
  }
}

// ---------------------------------------------------------------------------
// Gather (flip-odd permutation), add sinusoidal posenc, apply mamba LN.
// dir 0: row scan with odd-row reversal; dir 1: column scan w/ odd-col reversal.
// Output token-major (B,L,C).
// ---------------------------------------------------------------------------
__global__ void gather_pe_ln_kernel(const float* __restrict__ xn,
                                    const float* __restrict__ w,
                                    const float* __restrict__ b,
                                    float* __restrict__ out, int dir) {
  int i = blockIdx.x * blockDim.x + threadIdx.x;
  if (i >= Bb * LL) return;
  int bb = i / LL, l = i % LL;
  int ph, pw;
  if (dir == 0) { int h = l / Ww, ww = l % Ww; ph = h; pw = (h & 1) ? (Ww-1-ww) : ww; }
  else          { int ww = l / Hh, h = l % Hh; pw = ww; ph = (ww & 1) ? (Hh-1-h)  : h;  }
  float v[Cc]; float mu = 0.f;
#pragma unroll
  for (int c = 0; c < Cc; ++c) {
    int pair = c >> 1;
    float e = __expf(-9.210340371976184f * (float)(2*pair) / (float)Cc);
    float ang = (float)l * e;
    float pe = (c & 1) ? __cosf(ang) : __sinf(ang);
    v[c] = xn[((size_t)bb*Cc + c)*LL + (ph*Ww + pw)] + pe;
    mu += v[c];
  }
  mu *= (1.f / Cc);
  float var = 0.f;
#pragma unroll
  for (int c = 0; c < Cc; ++c) { float d = v[c] - mu; var += d*d; }
  var *= (1.f / Cc);
  float rs = rsqrtf(var + 1e-5f);
#pragma unroll
  for (int c = 0; c < Cc; ++c)
    out[((size_t)bb*LL + l)*Cc + c] = (v[c] - mu) * rs * w[c] + b[c];
}

// Causal depthwise conv1d (k=4, left pad 3) + SiLU.  xz token-major (B,L,192).
__global__ void dwconv_silu_kernel(const float* __restrict__ xz,
                                   const float* __restrict__ cw,
                                   const float* __restrict__ cb,
                                   float* __restrict__ xc) {
  int i = blockIdx.x * blockDim.x + threadIdx.x;
  if (i >= Bb * LL * DIv) return;
  int d = i % DIv; int t = i / DIv; int l = t % LL; int bb = t / LL;
  float acc = cb[d];
#pragma unroll
  for (int j = 0; j < DCONVv; ++j) {
    int ls = l - (DCONVv - 1) + j;
    if (ls >= 0) acc += cw[d*DCONVv + j] * xz[((size_t)bb*LL + ls)*(2*DIv) + d];
  }
  xc[(size_t)t*DIv + d] = acc / (1.f + __expf(-acc));
}

// dt = softplus(x_dbl[:, :3] @ dtw^T + dtb)
__global__ void dt_kernel(const float* __restrict__ xdbl,
                          const float* __restrict__ dtw,
                          const float* __restrict__ dtb,
                          float* __restrict__ dt) {
  int i = blockIdx.x * blockDim.x + threadIdx.x;
  if (i >= Bb * LL * DIv) return;
  int d = i % DIv; int t = i / DIv;
  float s = dtb[d];
#pragma unroll
  for (int r = 0; r < DTRv; ++r) s += xdbl[(size_t)t*XPN + r] * dtw[d*DTRv + r];
  dt[(size_t)t*DIv + d] = (s > 20.f) ? s : log1pf(__expf(s));
}

// Selective scan: 16 lanes per (b,d) chain (lane = state s), sequential in L.
// Cross-state reduction via __shfl_xor within 16-lane groups of the wave32.
__global__ void scan_kernel(const float* __restrict__ dtv,
                            const float* __restrict__ xdbl,
                            const float* __restrict__ xc,
                            const float* __restrict__ Alog,
                            float* __restrict__ ysc) {
  int gid = blockIdx.x * blockDim.x + threadIdx.x;
  if (gid >= Bb * DIv * DSv) return;
  int s  = gid & (DSv - 1);
  int d  = (gid >> 4) % DIv;
  int bb = gid / (DSv * DIv);
  float A = -__expf(Alog[d*DSv + s]);
  float h = 0.f;
  for (int l = 0; l < LL; ++l) {
    size_t t = (size_t)bb*LL + l;
    float dt = dtv[t*DIv + d];
    float xv = xc[t*DIv + d];
    float Bv = xdbl[t*XPN + DTRv + s];
    float Cv = xdbl[t*XPN + DTRv + DSv + s];
    h = __expf(dt * A) * h + dt * Bv * xv;
    float p = h * Cv;
    p += __shfl_xor(p, 8, 16);
    p += __shfl_xor(p, 4, 16);
    p += __shfl_xor(p, 2, 16);
    p += __shfl_xor(p, 1, 16);
    if (s == 0) ysc[t*DIv + d] = p;
  }
}

// y = (y_scan + xc*D) * silu(z)
__global__ void yact_kernel(const float* __restrict__ ysc,
                            const float* __restrict__ xc,
                            const float* __restrict__ xz,
                            const float* __restrict__ Dp,
                            float* __restrict__ yact) {
  int i = blockIdx.x * blockDim.x + threadIdx.x;
  if (i >= Bb * LL * DIv) return;
  int d = i % DIv; size_t t = i / DIv;
  float z = xz[t*(2*DIv) + DIv + d];
  float y = ysc[t*DIv + d] + xc[t*DIv + d] * Dp[d];
  yact[t*DIv + d] = y * (z / (1.f + __expf(-z)));
}

// x_mid = x + unflip(out1) + untranspose(unflip(out2))
__global__ void combine_kernel(const float* __restrict__ x,
                               const float* __restrict__ o1,
                               const float* __restrict__ o2,
                               float* __restrict__ xm) {
  int i = blockIdx.x * blockDim.x + threadIdx.x;
  if (i >= Bb * Cc * LL) return;
  int w = i % Ww; int h = (i / Ww) % Hh; int c = (i / LL) % Cc; int bb = i / (LL * Cc);
  int w1 = (h & 1) ? (Ww-1-w) : w;
  int h1 = (w & 1) ? (Hh-1-h) : h;
  float a = o1[((size_t)bb*LL + h*Ww + w1)*Cc + c];
  float b = o2[((size_t)bb*LL + w*Hh + h1)*Cc + c];
  xm[i] = x[i] + a + b;
}

// 2x2 average pool (B,48,64,64) -> (B,48,32,32)
__global__ void pool_kernel(const float* __restrict__ x, float* __restrict__ y) {
  int i = blockIdx.x * blockDim.x + threadIdx.x;
  if (i >= Bb * Cc * 32 * 32) return;
  int ww = i % 32; int hh = (i / 32) % 32; int c = (i / 1024) % Cc; int bb = i / (1024 * Cc);
  const float* p = x + ((size_t)bb*Cc + c)*LL;
  float s = p[(2*hh)*Ww + 2*ww] + p[(2*hh)*Ww + 2*ww+1]
          + p[(2*hh+1)*Ww + 2*ww] + p[(2*hh+1)*Ww + 2*ww+1];
  y[i] = 0.25f * s;
}

// 3x3 conv 48->48, pad 1, on (B,48,32,32)
__global__ void conv3x3_kernel(const float* __restrict__ in,
                               const float* __restrict__ w,
                               const float* __restrict__ bias,
                               float* __restrict__ out) {
  int i = blockIdx.x * blockDim.x + threadIdx.x;
  if (i >= Bb * Cc * 32 * 32) return;
  int ww = i % 32; int hh = (i / 32) % 32; int co = (i / 1024) % Cc; int bb = i / (1024 * Cc);
  float acc = bias[co];
  for (int ci = 0; ci < Cc; ++ci) {
    const float* p = in + ((size_t)bb*Cc + ci)*1024;
    const float* wp = w + ((size_t)co*Cc + ci)*9;
#pragma unroll
    for (int kh = 0; kh < 3; ++kh) {
      int h2 = hh + kh - 1; if (h2 < 0 || h2 >= 32) continue;
#pragma unroll
      for (int kw = 0; kw < 3; ++kw) {
        int w2 = ww + kw - 1; if (w2 < 0 || w2 >= 32) continue;
        acc += p[h2*32 + w2] * wp[kh*3 + kw];
      }
    }
  }
  out[i] = acc;
}

// Per-pixel channel LN + ReLU on (B,48,32,32)
__global__ void ln_relu32_kernel(const float* __restrict__ x,
                                 const float* __restrict__ w,
                                 const float* __restrict__ b,
                                 float* __restrict__ out) {
  int i = blockIdx.x * blockDim.x + threadIdx.x;
  if (i >= Bb * 1024) return;
  int bb = i / 1024, p = i % 1024;
  float v[Cc]; float mu = 0.f;
#pragma unroll
  for (int c = 0; c < Cc; ++c) { v[c] = x[((size_t)bb*Cc + c)*1024 + p]; mu += v[c]; }
  mu *= (1.f / Cc);
  float var = 0.f;
#pragma unroll
  for (int c = 0; c < Cc; ++c) { float d = v[c] - mu; var += d*d; }
  var *= (1.f / Cc);
  float rs = rsqrtf(var + 1e-5f);
#pragma unroll
  for (int c = 0; c < Cc; ++c) {
    float o = (v[c] - mu) * rs * w[c] + b[c];
    out[((size_t)bb*Cc + c)*1024 + p] = fmaxf(o, 0.f);
  }
}

// Depthwise 3x3, pad 1, token-major (B,L,CH)
__global__ void dw3x3_kernel(const float* __restrict__ in,
                             const float* __restrict__ wgt,
                             float* __restrict__ out, int CH) {
  int i = blockIdx.x * blockDim.x + threadIdx.x;
  if (i >= Bb * LL * CH) return;
  int c = i % CH; int t = i / CH; int l = t % LL; int bb = t / LL;
  int h = l / Ww, w = l % Ww;
  float acc = 0.f;
#pragma unroll
  for (int kh = 0; kh < 3; ++kh) {
    int h2 = h + kh - 1; if (h2 < 0 || h2 >= Hh) continue;
#pragma unroll
    for (int kw = 0; kw < 3; ++kw) {
      int w2 = w + kw - 1; if (w2 < 0 || w2 >= Ww) continue;
      acc += in[((size_t)bb*LL + h2*Ww + w2)*CH + c] * wgt[c*9 + kh*3 + kw];
    }
  }
  out[(size_t)t*CH + c] = acc;
}

// fin[:, :96] = xd[:, :96]; fin[:, 96:144] = nearest-upsampled pooled branch
__global__ void pack_fuse_kernel(const float* __restrict__ xd,
                                 const float* __restrict__ t2,
                                 float* __restrict__ fin) {
  int i = blockIdx.x * blockDim.x + threadIdx.x;
  const int FC = HFv + Cc; // 144
  if (i >= Bb * LL * FC) return;
  int ch = i % FC; int t = i / FC; int l = t % LL; int bb = t / LL;
  float v;
  if (ch < HFv) v = xd[(size_t)t*(2*HFv) + ch];
  else {
    int c = ch - HFv; int h = l / Ww, w = l % Ww;
    v = t2[((size_t)bb*Cc + c)*1024 + (h >> 1)*32 + (w >> 1)];
  }
  fin[(size_t)t*FC + ch] = v;
}

// gx = gelu_exact(x1dwa) * xd[:, 96:192]
__global__ void gelu_mul_kernel(const float* __restrict__ x1,
                                const float* __restrict__ xd,
                                float* __restrict__ gx) {
  int i = blockIdx.x * blockDim.x + threadIdx.x;
  if (i >= Bb * LL * HFv) return;
  int c = i % HFv; size_t t = i / HFv;
  float a = x1[t*HFv + c];
  float g = 0.5f * a * (1.f + erff(a * 0.70710678118654752f));
  gx[t*HFv + c] = g * xd[t*(2*HFv) + HFv + c];
}

// out = x_mid + ff (token-major -> spatial)
__global__ void final_add_kernel(const float* __restrict__ xm,
                                 const float* __restrict__ ff,
                                 float* __restrict__ out) {
  int i = blockIdx.x * blockDim.x + threadIdx.x;
  if (i >= Bb * Cc * LL) return;
  int w = i % Ww; int h = (i / Ww) % Hh; int c = (i / LL) % Cc; int bb = i / (LL * Cc);
  out[i] = xm[i] + ff[((size_t)bb*LL + h*Ww + w)*Cc + c];
}

// ---------------------------------------------------------------------------
static inline float* bump(char*& p, size_t nflt) {
  float* r = (float*)p; p += nflt * sizeof(float); return r;
}

extern "C" void kernel_launch(void* const* d_in, const int* in_sizes, int n_in,
                              void* d_out, int out_size, void* d_ws, size_t ws_size,
                              hipStream_t stream) {
  (void)in_sizes; (void)n_in; (void)out_size; (void)ws_size;
  const float* x      = (const float*)d_in[0];
  const float* n1_w   = (const float*)d_in[1];
  const float* n1_b   = (const float*)d_in[2];
  const float* n2_w   = (const float*)d_in[3];
  const float* n2_b   = (const float*)d_in[4];
  const float* mln_w[2] = { (const float*)d_in[5], (const float*)d_in[7] };
  const float* mln_b[2] = { (const float*)d_in[6], (const float*)d_in[8] };
  const float* fln1_w = (const float*)d_in[9];
  const float* fln1_b = (const float*)d_in[10];
  const float* fln2_w = (const float*)d_in[11];
  const float* fln2_b = (const float*)d_in[12];
  const float *inproj[2], *convw[2], *convb[2], *xproj[2], *dtw[2], *dtb[2],
              *Alog[2], *Dp[2], *outproj[2];
  for (int m = 0; m < 2; ++m) {
    int base = 13 + 9*m;
    inproj[m]  = (const float*)d_in[base+0];
    convw[m]   = (const float*)d_in[base+1];
    convb[m]   = (const float*)d_in[base+2];
    xproj[m]   = (const float*)d_in[base+3];
    dtw[m]     = (const float*)d_in[base+4];
    dtb[m]     = (const float*)d_in[base+5];
    Alog[m]    = (const float*)d_in[base+6];
    Dp[m]      = (const float*)d_in[base+7];
    outproj[m] = (const float*)d_in[base+8];
  }
  const float* f_pin = (const float*)d_in[31];
  const float* f_dw  = (const float*)d_in[32];
  const float* f_fus = (const float*)d_in[33];
  const float* f_dwa = (const float*)d_in[34];
  const float* f_pout= (const float*)d_in[35];
  const float* f_c1w = (const float*)d_in[36];
  const float* f_c1b = (const float*)d_in[37];
  const float* f_c2w = (const float*)d_in[38];
  const float* f_c2b = (const float*)d_in[39];
  float* out = (float*)d_out;

  // workspace layout
  char* p = (char*)d_ws;
  const size_t NT = (size_t)Bb * LL;               // 8192 tokens
  float* xn    = bump(p, NT * Cc);                 // LN'd spatial
  float* xln[2], *xz[2], *xc[2], *xdbl[2], *dtv[2], *ysc[2], *yact[2], *outp[2];
  for (int m = 0; m < 2; ++m) {
    xln[m]  = bump(p, NT * Cc);
    xz[m]   = bump(p, NT * 2 * DIv);
    xc[m]   = bump(p, NT * DIv);
    xdbl[m] = bump(p, NT * XPN);
    dtv[m]  = bump(p, NT * DIv);
    ysc[m]  = bump(p, NT * DIv);
    yact[m] = bump(p, NT * DIv);
    outp[m] = bump(p, NT * Cc);
  }
  float* xmid  = bump(p, NT * Cc);
  float* xnt   = bump(p, NT * Cc);
  float* xp    = bump(p, NT * 2 * HFv);
  float* xd    = bump(p, NT * 2 * HFv);
  float* y0    = bump(p, (size_t)Bb * Cc * 1024);
  float* c1t   = bump(p, (size_t)Bb * Cc * 1024);
  float* t1    = bump(p, (size_t)Bb * Cc * 1024);
  float* c2t   = bump(p, (size_t)Bb * Cc * 1024);
  float* t2    = bump(p, (size_t)Bb * Cc * 1024);
  float* fin   = bump(p, NT * (HFv + Cc));
  float* x1f   = bump(p, NT * HFv);
  float* x1d   = bump(p, NT * HFv);
  float* gx    = bump(p, NT * HFv);
  float* ff    = bump(p, NT * Cc);

  const int TB = 256;
  const int M  = (int)NT;           // 8192 rows, always multiple of 16
  const int MT = M / 16;            // 512 M-tiles

#define GEMM_LAUNCH(NN, KK, Aptr, Wptr, Cptr)                                  \
  do {                                                                         \
    int tn_ = ((NN) + 15) / 16;                                                \
    int waves_ = MT * tn_;                                                     \
    wmma_gemm_nt<NN, KK><<<CDIV(waves_ * 32, TB), TB, 0, stream>>>(            \
        (Aptr), (Wptr), (Cptr), M);                                            \
  } while (0)

  // --- Mamba branch prep ---
  ln4d_kernel<<<CDIV(M, TB), TB, 0, stream>>>(x, n1_w, n1_b, xn, 0);
  for (int m = 0; m < 2; ++m)
    gather_pe_ln_kernel<<<CDIV(M, TB), TB, 0, stream>>>(xn, mln_w[m], mln_b[m], xln[m], m);

  for (int m = 0; m < 2; ++m) {
    GEMM_LAUNCH(2*DIv, Cc, xln[m], inproj[m], xz[m]);                     // in-proj
    dwconv_silu_kernel<<<CDIV(M*DIv, TB), TB, 0, stream>>>(xz[m], convw[m], convb[m], xc[m]);
    GEMM_LAUNCH(XPN, DIv, xc[m], xproj[m], xdbl[m]);                      // x-proj
    dt_kernel<<<CDIV(M*DIv, TB), TB, 0, stream>>>(xdbl[m], dtw[m], dtb[m], dtv[m]);
    scan_kernel<<<CDIV(Bb*DIv*DSv, TB), TB, 0, stream>>>(dtv[m], xdbl[m], xc[m], Alog[m], ysc[m]);
    yact_kernel<<<CDIV(M*DIv, TB), TB, 0, stream>>>(ysc[m], xc[m], xz[m], Dp[m], yact[m]);
    GEMM_LAUNCH(Cc, DIv, yact[m], outproj[m], outp[m]);                   // out-proj
  }
  combine_kernel<<<CDIV(M*Cc, TB), TB, 0, stream>>>(x, outp[0], outp[1], xmid);

  // --- Conv-FFN branch ---
  ln4d_kernel<<<CDIV(M, TB), TB, 0, stream>>>(xmid, n2_w, n2_b, xnt, 1);
  GEMM_LAUNCH(2*HFv, Cc, xnt, f_pin, xp);                                 // pin 1x1
  dw3x3_kernel<<<CDIV(M*2*HFv, TB), TB, 0, stream>>>(xp, f_dw, xd, 2*HFv);

  pool_kernel<<<CDIV(Bb*Cc*1024, TB), TB, 0, stream>>>(x, y0);
  conv3x3_kernel<<<CDIV(Bb*Cc*1024, TB), TB, 0, stream>>>(y0, f_c1w, f_c1b, c1t);
  ln_relu32_kernel<<<CDIV(Bb*1024, TB), TB, 0, stream>>>(c1t, fln1_w, fln1_b, t1);
  conv3x3_kernel<<<CDIV(Bb*Cc*1024, TB), TB, 0, stream>>>(t1, f_c2w, f_c2b, c2t);
  ln_relu32_kernel<<<CDIV(Bb*1024, TB), TB, 0, stream>>>(c2t, fln2_w, fln2_b, t2);

  pack_fuse_kernel<<<CDIV(M*(HFv+Cc), TB), TB, 0, stream>>>(xd, t2, fin);
  GEMM_LAUNCH(HFv, HFv + Cc, fin, f_fus, x1f);                            // fus 1x1
  dw3x3_kernel<<<CDIV(M*HFv, TB), TB, 0, stream>>>(x1f, f_dwa, x1d, HFv);
  gelu_mul_kernel<<<CDIV(M*HFv, TB), TB, 0, stream>>>(x1d, xd, gx);
  GEMM_LAUNCH(Cc, HFv, gx, f_pout, ff);                                   // pout 1x1
  final_add_kernel<<<CDIV(M*Cc, TB), TB, 0, stream>>>(xmid, ff, out);

#undef GEMM_LAUNCH
}
